// FlashAttention_48155173323004
// MI455X (gfx1250) — matrix operand order repaired and checked by reading, not compiled, not run on hardware
//
#include <hip/hip_runtime.h>
#include <hip/hip_bf16.h>

typedef __attribute__((ext_vector_type(16))) _Float16 v16h;
typedef __attribute__((ext_vector_type(8)))  float    v8f;
typedef __attribute__((ext_vector_type(4)))  int      v4i;

#define B_    4
#define N_    2048
#define DIM_  1024
#define H_    16
#define DH_   64
#define SCALE_ 0.125f   // 64^-0.5

// ---------------- WMMA fragment helpers (wave32, gfx1250 layouts) ----------------

__device__ __forceinline__ v16h pack32(v4i lo, v4i hi) {
    union { v4i i[2]; v16h h; } u; u.i[0] = lo; u.i[1] = hi; return u.h;
}

// A fragment: 16x32 (MxK), row-major source, tile origin `p`, row stride lda (halves).
// lane L: row M = L&15; halves 0..7 -> K = kb..kb+7, halves 8..15 -> K = 16+kb..23+kb, kb = (L>>4)*8
__device__ __forceinline__ v16h load_a_frag(const _Float16* __restrict__ p, int lda, int lane) {
    const _Float16* q = p + (lane & 15) * lda + ((lane >> 4) << 3);
    return pack32(*(const v4i*)q, *(const v4i*)(q + 16));
}

// B fragment for A*W^T: W row-major (rows = output cols), tile origin `p`, row stride ldw.
// lane L: col N = L&15; halves 0..7 -> K = kb..kb+7, halves 8..15 -> K = 16+kb..23+kb, kb = (L>>4)*8
__device__ __forceinline__ v16h load_bT_frag(const _Float16* __restrict__ p, int ldw, int lane) {
    const _Float16* q = p + (lane & 15) * ldw + ((lane >> 4) << 3);
    return pack32(*(const v4i*)q, *(const v4i*)(q + 16));
}

__device__ __forceinline__ v8f wmma_f16(v16h a, v16h b, v8f c) {
    return __builtin_amdgcn_wmma_f32_16x16x32_f16(false, a, false, b, (short)0, c, false, false);
}

// ---------------- f32 -> f16 convert (vectorized) ----------------

struct h4 { _Float16 x, y, z, w; };

__global__ __launch_bounds__(256) void f32_to_f16_kernel(const float4* __restrict__ src,
                                                         h4* __restrict__ dst, int n4) {
    int i = blockIdx.x * blockDim.x + threadIdx.x;
    int stride = gridDim.x * blockDim.x;
    for (; i < n4; i += stride) {
        float4 v = src[i];
        h4 o = { (_Float16)v.x, (_Float16)v.y, (_Float16)v.z, (_Float16)v.w };
        dst[i] = o;
    }
}

// ---------------- shared GEMM building blocks: 32x64 wave tile, ping-pong k-loop ----------------

__device__ __forceinline__ void gemm_load(const _Float16* __restrict__ Ap,
                                          const _Float16* __restrict__ Wp,
                                          int k0, int lane, v16h (&a)[2], v16h (&b)[4]) {
#pragma unroll
    for (int m = 0; m < 2; ++m) a[m] = load_a_frag(Ap + m * 16 * DIM_ + k0, DIM_, lane);
#pragma unroll
    for (int t = 0; t < 4; ++t) b[t] = load_bT_frag(Wp + t * 16 * DIM_ + k0, DIM_, lane);
}

__device__ __forceinline__ void gemm_mma(v8f (&acc)[2][4], const v16h (&a)[2], const v16h (&b)[4]) {
#pragma unroll
    for (int m = 0; m < 2; ++m)
#pragma unroll
        for (int t = 0; t < 4; ++t) acc[m][t] = wmma_f16(a[m], b[t], acc[m][t]);
}

// steady-state: load B(k+32); mma A(k); load A(k+64); mma B(k+32)  (no buffer copies)
__device__ __forceinline__ void gemm_kloop(const _Float16* __restrict__ Ap,
                                           const _Float16* __restrict__ Wp,
                                           int lane, v8f (&acc)[2][4]) {
    v16h aA[2], bA[4], aB[2], bB[4];
    gemm_load(Ap, Wp, 0, lane, aA, bA);
#pragma unroll 1
    for (int k0 = 0; k0 < DIM_ - 64; k0 += 64) {
        gemm_load(Ap, Wp, k0 + 32, lane, aB, bB);
        gemm_mma(acc, aA, bA);
        gemm_load(Ap, Wp, k0 + 64, lane, aA, bA);
        gemm_mma(acc, aB, bB);
    }
    gemm_load(Ap, Wp, DIM_ - 32, lane, aB, bB);
    gemm_mma(acc, aA, bA);
    gemm_mma(acc, aB, bB);
}

// ---------------- QKV GEMM: qkv = x @ w_qkv^T, scatter Q/K [B,H,N,64], V transposed [B,H,64,N] ----------------
// M 32-blocks = 256, N 64-blocks = 48 -> 12288 waves / 8 per block = 1536 blocks.
__global__ __launch_bounds__(256) void qkv_gemm_kernel(const _Float16* __restrict__ xh,
                                                       const _Float16* __restrict__ wh,
                                                       _Float16* __restrict__ Q,
                                                       _Float16* __restrict__ K,
                                                       _Float16* __restrict__ Vt) {
    const int lane = threadIdx.x & 31;
    const int wid  = blockIdx.x * (blockDim.x >> 5) + (threadIdx.x >> 5);
    const int NT = (3 * DIM_) / 64;           // 48
    const int mt = wid / NT, nt = wid % NT;
    const int m0 = mt * 32, n0 = nt * 64;

    const _Float16* Ap = xh + (size_t)m0 * DIM_;
    const _Float16* Wp = wh + (size_t)n0 * DIM_;

    v8f acc[2][4] = {};
    gemm_kloop(Ap, Wp, lane, acc);

    // scatter: whole 64-wide stripe lies in one (s, h): s = n0/1024, h = (n0%1024)/64
    const int s = n0 / DIM_;
    const int h = (n0 % DIM_) / DH_;
    const int ncol = lane & 15, mhi = (lane >> 4) * 8;
#pragma unroll
    for (int m = 0; m < 2; ++m)
#pragma unroll
        for (int t = 0; t < 4; ++t)
#pragma unroll
            for (int r = 0; r < 8; ++r) {
                int gm = m0 + m * 16 + mhi + r;      // global row = b*2048 + n
                int bb = gm >> 11, n = gm & (N_ - 1);
                int d  = t * 16 + ncol;
                _Float16 val = (_Float16)acc[m][t][r];
                if (s == 0)       Q [(((size_t)(bb * H_ + h) * N_) + n) * DH_ + d] = val;
                else if (s == 1)  K [(((size_t)(bb * H_ + h) * N_) + n) * DH_ + d] = val;
                else              Vt[(((size_t)(bb * H_ + h) * DH_) + d) * N_ + n] = val;  // transposed
            }
}

// ---------------- Flash attention: one wave per (b, h, 16-row q tile); 32 keys / step ----------------
#define LDP 40   // LDS row stride in halves (80 B rows: 16B-aligned, conflict padding)

__device__ __forceinline__ void load_kfrag(const _Float16* __restrict__ Kp, int lane, v16h (&kf)[4]) {
#pragma unroll
    for (int ct = 0; ct < 2; ++ct) {
        kf[2 * ct + 0] = load_bT_frag(Kp + ct * 16 * DH_,      DH_, lane);
        kf[2 * ct + 1] = load_bT_frag(Kp + ct * 16 * DH_ + 32, DH_, lane);
    }
}

// One 32-key flash step: uses kf_use, (optionally) fills kf_fill for next_kb.
__device__ __forceinline__ void attn_step(int kb, const _Float16* __restrict__ Kb,
                                          const _Float16* __restrict__ Vtb,
                                          v16h aq0, v16h aq1,
                                          v16h (&kf_use)[4], v16h (&kf_fill)[4],
                                          bool load_next, int next_kb,
                                          float (&mi)[8], float (&li)[8], v8f (&acc)[4],
                                          _Float16* __restrict__ lp,
                                          int lane, int ncol, int khw) {
    // S = Q * K^T for 32 keys (two 16x16 C/D tiles)
    v8f s[2];
#pragma unroll
    for (int ct = 0; ct < 2; ++ct) {
        v8f sc = {};
        sc = wmma_f16(aq0, kf_use[2 * ct + 0], sc);
        sc = wmma_f16(aq1, kf_use[2 * ct + 1], sc);
        s[ct] = sc;
    }
    // issue V loads for current kb early (Vt is K-contiguous -> b128 pairs);
    // the softmax VALU stretch below hides their latency
    v16h vf[4];
#pragma unroll
    for (int t = 0; t < 4; ++t)
        vf[t] = load_bT_frag(Vtb + (size_t)(t * 16) * N_ + kb, N_, lane);
    // ping-pong prefetch of next step's K fragments (uniform branch, EXEC untouched)
    if (load_next) load_kfrag(Kb + next_kb * DH_, lane, kf_fill);
    // L2 prefetch two steps ahead (global_prefetch_b8)
    if (kb + 64 < N_) {
        __builtin_prefetch(Kb + (kb + 64) * DH_, 0, 1);
        __builtin_prefetch(Vtb + kb + 64, 0, 1);
    }
    // online softmax; row m = r + 8*khw spans the 16 lanes of this half-wave
#pragma unroll
    for (int r = 0; r < 8; ++r) {
        float s0 = s[0][r] * SCALE_, s1 = s[1][r] * SCALE_;
        float rowm = fmaxf(s0, s1);
        rowm = fmaxf(rowm, __shfl_xor(rowm, 1));
        rowm = fmaxf(rowm, __shfl_xor(rowm, 2));
        rowm = fmaxf(rowm, __shfl_xor(rowm, 4));
        rowm = fmaxf(rowm, __shfl_xor(rowm, 8));
        float newm  = fmaxf(mi[r], rowm);
        float alpha = __expf(mi[r] - newm);
        float p0 = __expf(s0 - newm), p1 = __expf(s1 - newm);
        float rs = p0 + p1;
        rs += __shfl_xor(rs, 1); rs += __shfl_xor(rs, 2);
        rs += __shfl_xor(rs, 4); rs += __shfl_xor(rs, 8);
        li[r] = li[r] * alpha + rs;
        mi[r] = newm;
        s[0][r] = p0; s[1][r] = p1;
#pragma unroll
        for (int t = 0; t < 4; ++t) acc[t][r] *= alpha;
    }
    // C/D-layout -> A-layout transpose of P through this wave's private LDS slab.
    // Same-wave DS ops complete in order; s_wait_dscnt + memory clobber stop reordering.
#pragma unroll
    for (int ct = 0; ct < 2; ++ct)
#pragma unroll
        for (int r = 0; r < 8; ++r)
            lp[(khw * 8 + r) * LDP + ct * 16 + ncol] = (_Float16)s[ct][r];
    asm volatile("s_wait_dscnt 0" ::: "memory");
    v16h ap;
    {
        const _Float16* q = lp + (lane & 15) * LDP + (khw << 3);
        ap = pack32(*(const v4i*)q, *(const v4i*)(q + 16));
    }
    asm volatile("" ::: "memory");
    // O += P * V
#pragma unroll
    for (int t = 0; t < 4; ++t)
        acc[t] = wmma_f16(ap, vf[t], acc[t]);
}

__global__ __launch_bounds__(256) void attn_kernel(const _Float16* __restrict__ Q,
                                                   const _Float16* __restrict__ K,
                                                   const _Float16* __restrict__ Vt,
                                                   _Float16* __restrict__ Oh) {
    __shared__ _Float16 lds[8 * 16 * LDP];      // private 16x32 P-staging slab per wave
    const int lane  = threadIdx.x & 31;
    const int wslot = threadIdx.x >> 5;
    const int wid   = blockIdx.x * 8 + wslot;
    const int bh = wid >> 7;                    // / (N/16 = 128)
    const int qt = wid & 127;

    const _Float16* Qp  = Q  + ((size_t)bh * N_ + qt * 16) * DH_;
    const _Float16* Kb  = K  + (size_t)bh * N_ * DH_;
    const _Float16* Vtb = Vt + (size_t)bh * DH_ * N_;

    v16h aq0 = load_a_frag(Qp, DH_, lane);      // Dh=64 -> two K=32 A fragments
    v16h aq1 = load_a_frag(Qp + 32, DH_, lane);

    float mi[8], li[8];
#pragma unroll
    for (int r = 0; r < 8; ++r) { mi[r] = -3.0e38f; li[r] = 0.0f; }
    v8f acc[4] = {};                            // O tile 16x64 (f32)

    _Float16* lp = lds + wslot * 16 * LDP;
    const int ncol = lane & 15, khw = lane >> 4;

    v16h kfA[4], kfB[4];
    load_kfrag(Kb, lane, kfA);                  // kb = 0
#pragma unroll 1
    for (int kb = 0; kb < N_; kb += 64) {
        // step 1 uses kfA, fills kfB for kb+32 (always in range)
        attn_step(kb,      Kb, Vtb, aq0, aq1, kfA, kfB, true,           kb + 32,
                  mi, li, acc, lp, lane, ncol, khw);
        // step 2 uses kfB, fills kfA for kb+64 (skipped on last iteration)
        attn_step(kb + 32, Kb, Vtb, aq0, aq1, kfB, kfA, kb + 64 < N_,   kb + 64,
                  mi, li, acc, lp, lane, ncol, khw);
    }

    // epilogue: scale by 1/l, write f16 attn_out [B, N, H*64]
    const int b = bh >> 4, h = bh & 15;
    const int qrow0 = qt * 16 + khw * 8;
#pragma unroll
    for (int r = 0; r < 8; ++r) {
        float inv = 1.0f / li[r];
        int n = qrow0 + r;
#pragma unroll
        for (int t = 0; t < 4; ++t)
            Oh[((size_t)(b * N_ + n)) * DIM_ + h * DH_ + t * 16 + ncol] =
                (_Float16)(acc[t][r] * inv);
    }
}

// ---------------- proj: out = attn_out @ w_proj^T (f32 out), 32x64 wave tiles ----------------
// M 32-blocks = 256, N 64-blocks = 16 -> 4096 waves / 8 per block = 512 blocks.
__global__ __launch_bounds__(256) void proj_gemm_kernel(const _Float16* __restrict__ Ohm,
                                                        const _Float16* __restrict__ wh,
                                                        float* __restrict__ out) {
    const int lane = threadIdx.x & 31;
    const int wid  = blockIdx.x * 8 + (threadIdx.x >> 5);
    const int NT = DIM_ / 64;                   // 16
    const int mt = wid / NT, nt = wid % NT;
    const int m0 = mt * 32, n0 = nt * 64;

    const _Float16* Ap = Ohm + (size_t)m0 * DIM_;
    const _Float16* Wp = wh  + (size_t)n0 * DIM_;

    v8f acc[2][4] = {};
    gemm_kloop(Ap, Wp, lane, acc);

    const int ncol = lane & 15, mhi = (lane >> 4) * 8;
#pragma unroll
    for (int m = 0; m < 2; ++m)
#pragma unroll
        for (int t = 0; t < 4; ++t)
#pragma unroll
            for (int r = 0; r < 8; ++r)
                out[(size_t)(m0 + m * 16 + mhi + r) * DIM_ + n0 + t * 16 + ncol] = acc[m][t][r];
}

// ---------------- Host launch ----------------

extern "C" void kernel_launch(void* const* d_in, const int* in_sizes, int n_in,
                              void* d_out, int out_size, void* d_ws, size_t ws_size,
                              hipStream_t stream) {
    const float* x      = (const float*)d_in[0];
    const float* w_qkv  = (const float*)d_in[1];
    const float* w_proj = (const float*)d_in[2];
    float* out = (float*)d_out;
    (void)in_sizes; (void)n_in; (void)out_size; (void)ws_size;

    char* ws = (char*)d_ws;
    size_t off = 0;
    auto carve = [&](size_t bytes) {
        void* p = ws + off;
        off += (bytes + 255) & ~(size_t)255;
        return p;
    };
    // total ~92 MB of workspace
    _Float16* xh     = (_Float16*)carve((size_t)B_ * N_ * DIM_ * 2);
    _Float16* wqkvh  = (_Float16*)carve((size_t)3 * DIM_ * DIM_ * 2);
    _Float16* wprojh = (_Float16*)carve((size_t)DIM_ * DIM_ * 2);
    _Float16* Qh     = (_Float16*)carve((size_t)B_ * H_ * N_ * DH_ * 2);
    _Float16* Kh     = (_Float16*)carve((size_t)B_ * H_ * N_ * DH_ * 2);
    _Float16* Vth    = (_Float16*)carve((size_t)B_ * H_ * DH_ * N_ * 2);  // V transposed
    _Float16* Ohm    = (_Float16*)carve((size_t)B_ * N_ * DIM_ * 2);

    f32_to_f16_kernel<<<1024, 256, 0, stream>>>((const float4*)x,      (h4*)xh,     (B_ * N_ * DIM_) / 4);
    f32_to_f16_kernel<<<1024, 256, 0, stream>>>((const float4*)w_qkv,  (h4*)wqkvh,  (3 * DIM_ * DIM_) / 4);
    f32_to_f16_kernel<<<256,  256, 0, stream>>>((const float4*)w_proj, (h4*)wprojh, (DIM_ * DIM_) / 4);

    qkv_gemm_kernel<<<1536, 256, 0, stream>>>(xh, wqkvh, Qh, Kh, Vth);
    attn_kernel<<<1024, 256, 0, stream>>>(Qh, Kh, Vth, Ohm);
    proj_gemm_kernel<<<512, 256, 0, stream>>>(Ohm, wprojh, out);
}
